// Decoder_44135083934179
// MI455X (gfx1250) — matrix operand order repaired and checked by reading, not compiled, not run on hardware
//
#include <hip/hip_runtime.h>

// ---------------------------------------------------------------------------
// MI455X (gfx1250) fused transformer decoder.
// Everything fits in the 192MB L2 (weights ~52MB as f16), so the plan is:
//   - once per launch: cast+transpose all weights to f16 [N][K] (WMMA B layout)
//   - all GEMMs via v_wmma_f32_16x16x32_f16, f32 accum, fused epilogues
//   - banded attention: one wave per (b, head, 16-query tile); 46-key band
//     padded to 48 -> scores/relbias/P·V all via WMMA, softmax in LDS.
// ---------------------------------------------------------------------------

#define TSEQ   512
#define DMODEL 512
#define NHEADS 8
#define DHEAD  64
#define DFF    2048
#define NROWS  2048   // B*T
#define NLAYER 6

typedef _Float16 h8v  __attribute__((ext_vector_type(8)));
typedef _Float16 h16v __attribute__((ext_vector_type(16)));
typedef float    f8v  __attribute__((ext_vector_type(8)));

#define WMMA_F16(a, b, c) \
  __builtin_amdgcn_wmma_f32_16x16x32_f16(false, (a), false, (b), (short)0, (c), false, false)

// A fragment (16x32 f16): lane holds row `row`, K chunks kc..kc+7 and kc+16..kc+23,
// kc = (lane>>4)*8. Two contiguous 16B loads from row-major storage.
__device__ __forceinline__ h16v frag_a(const _Float16* p, int row, int k0, int ld) {
  int l = threadIdx.x & 31;
  const _Float16* q = p + (size_t)row * ld + k0 + ((l >> 4) << 3);
  h8v lo = *(const h8v*)q;
  h8v hi = *(const h8v*)(q + 16);
  h16v f;
#pragma unroll
  for (int i = 0; i < 8; ++i) { f[i] = lo[i]; f[i + 8] = hi[i]; }
  return f;
}

// B fragment (32x16 f16): lane holds column `col` (caller passes the row index in the
// N-major transposed matrix Bt[N][K]); K chunks kc..kc+7 and kc+16..kc+23, kc = (lane>>4)*8.
__device__ __forceinline__ h16v frag_b(const _Float16* p, int col, int k0, int ld) {
  int l = threadIdx.x & 31;
  const _Float16* q = p + (size_t)col * ld + k0 + ((l >> 4) << 3);
  h8v lo = *(const h8v*)q;
  h8v hi = *(const h8v*)(q + 16);
  h16v f;
#pragma unroll
  for (int i = 0; i < 8; ++i) { f[i] = lo[i]; f[i + 8] = hi[i]; }
  return f;
}

// ---------------------------------------------------------------------------
// Generic WMMA GEMM: C[M,N] = act(A[M,K] @ W + bias) (+ residual)
// A: f16 row-major (lda), Bt: f16 [N][K] (ldb). 8 waves/block, 32x32 per wave,
// block tile 64(M) x 128(N). act: 0=none 1=prelu(alpha) 2=relu.
// ---------------------------------------------------------------------------
__global__ __launch_bounds__(256)
void k_gemm(const _Float16* __restrict__ A, const _Float16* __restrict__ Bt,
            const float* __restrict__ bias, float* __restrict__ Cf,
            _Float16* __restrict__ Ch, int M, int N, int K,
            int lda, int ldb, int ldc, int act, const float* alphaPtr, int addC)
{
  int l = threadIdx.x & 31;
  int wave = threadIdx.x >> 5;
  int wm = wave >> 2, wn = wave & 3;
  int m0 = blockIdx.y * 64 + wm * 32;
  int n0 = blockIdx.x * 128 + wn * 32;
  if (m0 >= M || n0 >= N) return;

  f8v acc[2][2] = {};
  int ar0 = m0 + (l & 15), ar1 = ar0 + 16;
  int br0 = n0 + (l & 15), br1 = br0 + 16;

  for (int k = 0; k < K; k += 32) {
    h16v a0 = frag_a(A, ar0, k, lda);
    h16v a1 = frag_a(A, ar1, k, lda);
    h16v b0 = frag_b(Bt, br0, k, ldb);
    h16v b1 = frag_b(Bt, br1, k, ldb);
    acc[0][0] = WMMA_F16(a0, b0, acc[0][0]);
    acc[0][1] = WMMA_F16(a0, b1, acc[0][1]);
    acc[1][0] = WMMA_F16(a1, b0, acc[1][0]);
    acc[1][1] = WMMA_F16(a1, b1, acc[1][1]);
  }

  float alpha = alphaPtr ? alphaPtr[0] : 0.f;
#pragma unroll
  for (int mi = 0; mi < 2; ++mi) {
#pragma unroll
    for (int ni = 0; ni < 2; ++ni) {
      int ncol = n0 + ni * 16 + (l & 15);
      int rbase = m0 + mi * 16 + ((l >> 4) << 3);
#pragma unroll
      for (int r = 0; r < 8; ++r) {
        float c = acc[mi][ni][r] + (bias ? bias[ncol] : 0.f);
        if (act == 1)      c = (c >= 0.f) ? c : alpha * c;
        else if (act == 2) c = fmaxf(c, 0.f);
        size_t idx = (size_t)(rbase + r) * ldc + ncol;
        if (addC) c += Cf[idx];
        if (Cf) Cf[idx] = c;
        if (Ch) Ch[idx] = (_Float16)c;
      }
    }
  }
}

// ---------------------------------------------------------------------------
// LayerNorm over 512 cols, f32 in -> f16 out. One wave per row, 8 rows/block.
// ---------------------------------------------------------------------------
__global__ __launch_bounds__(256)
void k_layernorm(const float* __restrict__ x, const float* __restrict__ g,
                 const float* __restrict__ b, _Float16* __restrict__ y, int M)
{
  int wave = threadIdx.x >> 5, l = threadIdx.x & 31;
  int row = blockIdx.x * 8 + wave;
  if (row >= M) return;
  const float* xr = x + (size_t)row * DMODEL;
  float v[16], s = 0.f;
#pragma unroll
  for (int i = 0; i < 16; ++i) { v[i] = xr[l + 32 * i]; s += v[i]; }
  for (int m = 16; m >= 1; m >>= 1) s += __shfl_xor(s, m, 32);
  float mu = s * (1.f / DMODEL);
  float s2 = 0.f;
#pragma unroll
  for (int i = 0; i < 16; ++i) { float d = v[i] - mu; s2 += d * d; }
  for (int m = 16; m >= 1; m >>= 1) s2 += __shfl_xor(s2, m, 32);
  float inv = rsqrtf(s2 * (1.f / DMODEL) + 1e-5f);
#pragma unroll
  for (int i = 0; i < 16; ++i) {
    int c = l + 32 * i;
    y[(size_t)row * DMODEL + c] = (_Float16)((v[i] - mu) * inv * g[c] + b[c]);
  }
}

// W f32 [K,N] -> Wt f16 [N,Kpad] (zero pad K..Kpad). 256 thr = 16x16 tile.
__global__ void k_transpose_cast(const float* __restrict__ W, _Float16* __restrict__ Wt,
                                 int K, int N, int Kpad)
{
  int n = blockIdx.x * 16 + (threadIdx.x & 15);
  int k = blockIdx.y * 16 + (threadIdx.x >> 4);
  if (n < N && k < Kpad)
    Wt[(size_t)n * Kpad + k] = (k < K) ? (_Float16)W[(size_t)k * N + n] : (_Float16)0.f;
}

// Build X0h [2048, 544] = f16(concat(motion*mask, mask, traj), zero pad 517..543)
__global__ void k_embed(const float* __restrict__ motion, const float* __restrict__ mask,
                        const float* __restrict__ traj, _Float16* __restrict__ X0h)
{
  int row = blockIdx.x;
  for (int c = threadIdx.x; c < 544; c += blockDim.x) {
    float v;
    if (c < 256)      v = motion[(size_t)row * 256 + c] * mask[(size_t)row * 256 + c];
    else if (c < 512) v = mask[(size_t)row * 256 + (c - 256)];
    else if (c < 517) v = traj[(size_t)row * 5 + (c - 512)];
    else              v = 0.f;
    X0h[(size_t)row * 544 + c] = (_Float16)v;
  }
}

// Relative-position lookup table: Lh [32][64] f16 (row 31 = 0 pad).
__global__ void k_rel_table(const float* __restrict__ w1, const float* __restrict__ b1,
                            const float* __restrict__ aPtr, const float* __restrict__ w2,
                            const float* __restrict__ b2, _Float16* __restrict__ Lh)
{
  float a = aPtr[0];
  for (int o = threadIdx.x; o < 32 * 64; o += blockDim.x) {
    int r = o >> 6, d = o & 63;
    float s = 0.f;
    if (r < 31) {
      float rp = (float)(r - 15);
      for (int c = 0; c < 512; ++c) {
        float t = rp * w1[c] + b1[c];
        t = (t >= 0.f) ? t : a * t;
        s += t * w2[(size_t)c * 64 + d];
      }
      s += b2[d];
    }
    Lh[o] = (_Float16)s;
  }
}

// ---------------------------------------------------------------------------
// Banded attention. One wave per (b, head, 16-query tile). Key band: 48 keys
// starting at j0 = i0-16 covers all |j-i|<=15. 18 WMMA ops per wave.
// ---------------------------------------------------------------------------
__global__ __launch_bounds__(32)
void k_attn(const _Float16* __restrict__ Qh, const _Float16* __restrict__ Kh,
            const _Float16* __restrict__ Vh, const _Float16* __restrict__ Lh,
            _Float16* __restrict__ AOh)
{
  __shared__ float S[16][48];
  __shared__ float REL[16][32];
  __shared__ __align__(16) _Float16 P[16][64];
  __shared__ __align__(16) _Float16 Vt[64][64];

  int l = threadIdx.x;
  int i0 = blockIdx.x * 16;
  int hh = blockIdx.y;
  int b  = blockIdx.z;
  int rb = b * TSEQ;       // row base in [NROWS] space
  int cc = hh * DHEAD;     // channel base
  int j0 = i0 - 16;

  h16v q0 = frag_a(Qh, rb + i0 + (l & 15), cc, DMODEL);
  h16v q1 = frag_a(Qh, rb + i0 + (l & 15), cc + 32, DMODEL);

  // scores S[q][key] = Q . K
  for (int nt = 0; nt < 3; ++nt) {
    int key = j0 + nt * 16 + (l & 15);
    key = key < 0 ? 0 : (key > TSEQ - 1 ? TSEQ - 1 : key);
    h16v kb0 = frag_b(Kh, rb + key, cc, DMODEL);
    h16v kb1 = frag_b(Kh, rb + key, cc + 32, DMODEL);
    f8v acc = {};
    acc = WMMA_F16(q0, kb0, acc);
    acc = WMMA_F16(q1, kb1, acc);
    int col = nt * 16 + (l & 15);
    int rbase = (l >> 4) << 3;
#pragma unroll
    for (int r = 0; r < 8; ++r) S[rbase + r][col] = acc[r];
  }

  // rel bias REL[q][r] = Q . lookup[r]
  for (int nt = 0; nt < 2; ++nt) {
    int rr = nt * 16 + (l & 15);           // rows 0..31 of Lh (31 = zero pad)
    h16v lb0 = frag_b(Lh, rr, 0, DHEAD);
    h16v lb1 = frag_b(Lh, rr, 32, DHEAD);
    f8v acc = {};
    acc = WMMA_F16(q0, lb0, acc);
    acc = WMMA_F16(q1, lb1, acc);
    int col = nt * 16 + (l & 15);
    int rbase = (l >> 4) << 3;
#pragma unroll
    for (int r = 0; r < 8; ++r) REL[rbase + r][col] = acc[r];
  }
  __syncthreads();

  // softmax (lanes 0..15, one row each)
  if (l < 16) {
    int i = i0 + l;
    float mx = -1e30f;
    for (int jj = 0; jj < 48; ++jj) {
      int j = j0 + jj, off = j - i;
      float v;
      if (j >= 0 && j < TSEQ && off >= -15 && off <= 15)
        v = (S[l][jj] + REL[l][off + 15]) * 0.125f;   // 1/sqrt(64)
      else
        v = -1e9f;
      S[l][jj] = v;
      mx = fmaxf(mx, v);
    }
    float sum = 0.f;
    for (int jj = 0; jj < 48; ++jj) {
      float e = __expf(S[l][jj] - mx);
      S[l][jj] = e;
      sum += e;
    }
    float inv = 1.f / sum;
    for (int jj = 0; jj < 64; ++jj)
      P[l][jj] = (jj < 48) ? (_Float16)(S[l][jj] * inv) : (_Float16)0.f;
  }

  // stage V transposed: Vt[d][key], keys 48..63 zeroed (P=0 there; keep finite)
  for (int idx = l; idx < 64 * 64; idx += 32) {
    int kk = idx >> 6, d = idx & 63;
    _Float16 v = (_Float16)0.f;
    if (kk < 48) {
      int j = j0 + kk;
      j = j < 0 ? 0 : (j > TSEQ - 1 ? TSEQ - 1 : j);
      v = Vh[(size_t)(rb + j) * DMODEL + cc + d];
    }
    Vt[d][kk] = v;
  }
  __syncthreads();

  // out[q][d] = P . V
  h16v pa0 = frag_a(&P[0][0], (l & 15), 0, 64);
  h16v pa1 = frag_a(&P[0][0], (l & 15), 32, 64);
  for (int nt = 0; nt < 4; ++nt) {
    h16v vb0 = frag_b(&Vt[0][0], nt * 16 + (l & 15), 0, 64);
    h16v vb1 = frag_b(&Vt[0][0], nt * 16 + (l & 15), 32, 64);
    f8v acc = {};
    acc = WMMA_F16(pa0, vb0, acc);
    acc = WMMA_F16(pa1, vb1, acc);
    int col = cc + nt * 16 + (l & 15);
    int rbase = i0 + ((l >> 4) << 3);
#pragma unroll
    for (int r = 0; r < 8; ++r)
      AOh[(size_t)(rb + rbase + r) * DMODEL + col] = (_Float16)acc[r];
  }
}

// out = dec*(1-mask) + motion*mask   (256 channels)
__global__ void k_mix(const float* __restrict__ dec, const float* __restrict__ motion,
                      const float* __restrict__ mask, float* __restrict__ out, int n)
{
  int i = blockIdx.x * blockDim.x + threadIdx.x;
  if (i < n) {
    float m = mask[i];
    out[i] = dec[i] * (1.f - m) + motion[i] * m;
  }
}

// ---------------------------------------------------------------------------
// Host orchestration
// ---------------------------------------------------------------------------
// d_in layout: 0 motion, 1 traj, 2 z, 3 mask, then params flattened by JAX
// pytree order (alphabetical dict keys at each level):
//   4 dec1.b, 5 dec1.w, 6 dec2.b, 7 dec2.w, 8 dec_a, 9 final_ln.b, 10 final_ln.g,
//   11 + 26*L + off for layer L, off:
//     0 cross.ln.b 1 cross.ln.g 2 cross.wk.b 3 cross.wk.w 4 cross.wo.b 5 cross.wo.w
//     6 cross.wq.b 7 cross.wq.w 8 cross.wv.b 9 cross.wv.w
//     10 pffn.l1.b 11 pffn.l1.w 12 pffn.l2.b 13 pffn.l2.w 14 pffn.ln.b 15 pffn.ln.g
//     16 self.ln.b 17 self.ln.g 18 self.wk.b 19 self.wk.w 20 self.wo.b 21 self.wo.w
//     22 self.wq.b 23 self.wq.w 24 self.wv.b 25 self.wv.w
//   167 me1.b, 168 me1.w, 169 me1_a, 170 me2.b, 171 me2.w, 172 me2_a,
//   173 rpe1.b, 174 rpe1.w, 175 rpe2.b, 176 rpe2.w, 177 rpe_a

extern "C" void kernel_launch(void* const* d_in, const int* in_sizes, int n_in,
                              void* d_out, int out_size, void* d_ws, size_t ws_size,
                              hipStream_t stream)
{
  (void)in_sizes; (void)n_in; (void)out_size; (void)ws_size;
  const float* motion = (const float*)d_in[0];
  const float* traj   = (const float*)d_in[1];
  const float* z      = (const float*)d_in[2];
  const float* maskp  = (const float*)d_in[3];
  auto F = [&](int i) { return (const float*)d_in[i]; };

  // workspace carve-up (256B aligned)
  size_t off = 0;
  char* base = (char*)d_ws;
  auto alloc = [&](size_t bytes) -> void* {
    void* p = base + off;
    off += (bytes + 255) & ~(size_t)255;
    return p;
  };
  _Float16* me1t  = (_Float16*)alloc((size_t)512 * 544 * 2);
  _Float16* me2t  = (_Float16*)alloc((size_t)512 * 512 * 2);
  _Float16* dec1t = (_Float16*)alloc((size_t)512 * 512 * 2);
  _Float16* dec2t = (_Float16*)alloc((size_t)256 * 512 * 2);
  // per-layer weight slab (reused each layer)
  _Float16* s_wqt = (_Float16*)alloc((size_t)512 * 512 * 2);
  _Float16* s_wkt = (_Float16*)alloc((size_t)512 * 512 * 2);
  _Float16* s_wvt = (_Float16*)alloc((size_t)512 * 512 * 2);
  _Float16* s_wot = (_Float16*)alloc((size_t)512 * 512 * 2);
  _Float16* c_wqt = (_Float16*)alloc((size_t)512 * 512 * 2);
  _Float16* c_wkt = (_Float16*)alloc((size_t)512 * 512 * 2);
  _Float16* c_wvt = (_Float16*)alloc((size_t)512 * 512 * 2);
  _Float16* c_wot = (_Float16*)alloc((size_t)512 * 512 * 2);
  _Float16* l1t   = (_Float16*)alloc((size_t)2048 * 512 * 2);
  _Float16* l2t   = (_Float16*)alloc((size_t)512 * 2048 * 2);
  // activations
  _Float16* Lh   = (_Float16*)alloc((size_t)32 * 64 * 2);
  _Float16* X0h  = (_Float16*)alloc((size_t)NROWS * 544 * 2);
  _Float16* H1h  = (_Float16*)alloc((size_t)NROWS * 512 * 2);
  float*    HBUF = (float*)   alloc((size_t)NROWS * 512 * 4);
  _Float16* XNh  = (_Float16*)alloc((size_t)NROWS * 512 * 2);
  _Float16* ZNh  = (_Float16*)alloc((size_t)NROWS * 512 * 2);
  _Float16* Qh   = (_Float16*)alloc((size_t)NROWS * 512 * 2);
  _Float16* Kh   = (_Float16*)alloc((size_t)NROWS * 512 * 2);
  _Float16* Vh   = (_Float16*)alloc((size_t)NROWS * 512 * 2);
  _Float16* AOh  = (_Float16*)alloc((size_t)NROWS * 512 * 2);
  _Float16* F1h  = (_Float16*)alloc((size_t)NROWS * 2048 * 2);
  _Float16* D1h  = (_Float16*)alloc((size_t)NROWS * 512 * 2);
  float*    DECf = (float*)   alloc((size_t)NROWS * 256 * 4);

  auto transpose = [&](const float* W, _Float16* Wt, int K, int N, int Kpad) {
    dim3 g((N + 15) / 16, (Kpad + 15) / 16);
    k_transpose_cast<<<g, 256, 0, stream>>>(W, Wt, K, N, Kpad);
  };
  auto gemm = [&](const _Float16* A, const _Float16* Bt, const float* bias,
                  float* Cf, _Float16* Ch, int M, int N, int K,
                  int lda, int ldb, int ldc, int act, const float* alpha, int addC) {
    dim3 g(N / 128, M / 64);
    k_gemm<<<g, 256, 0, stream>>>(A, Bt, bias, Cf, Ch, M, N, K, lda, ldb, ldc,
                                  act, alpha, addC);
  };
  auto lnorm = [&](const float* x, const float* g, const float* b, _Float16* y) {
    k_layernorm<<<NROWS / 8, 256, 0, stream>>>(x, g, b, y, NROWS);
  };

  // prologue: weight caches, rel table, embedding
  transpose(F(168), me1t, 517, 512, 544);
  transpose(F(171), me2t, 512, 512, 512);
  transpose(F(5),   dec1t, 512, 512, 512);
  transpose(F(7),   dec2t, 512, 256, 512);
  k_rel_table<<<1, 256, 0, stream>>>(F(174), F(173), F(177), F(176), F(175), Lh);
  k_embed<<<NROWS, 256, 0, stream>>>(motion, maskp, traj, X0h);

  // motion encoder MLP
  gemm(X0h, me1t, F(167), nullptr, H1h, NROWS, 512, 544, 544, 544, 512, 1, F(169), 0);
  gemm(H1h, me2t, F(170), HBUF, nullptr, NROWS, 512, 512, 512, 512, 512, 1, F(172), 0);

  dim3 agrid(TSEQ / 16, NHEADS, 4);
  for (int L = 0; L < NLAYER; ++L) {
    int lb = 11 + 26 * L;
    // stage this layer's weights into the f16 slab
    transpose(F(lb + 23), s_wqt, 512, 512, 512);
    transpose(F(lb + 19), s_wkt, 512, 512, 512);
    transpose(F(lb + 25), s_wvt, 512, 512, 512);
    transpose(F(lb + 21), s_wot, 512, 512, 512);
    transpose(F(lb + 7),  c_wqt, 512, 512, 512);
    transpose(F(lb + 3),  c_wkt, 512, 512, 512);
    transpose(F(lb + 9),  c_wvt, 512, 512, 512);
    transpose(F(lb + 5),  c_wot, 512, 512, 512);
    transpose(F(lb + 11), l1t, 512, 2048, 512);
    transpose(F(lb + 13), l2t, 2048, 512, 2048);

    // self attention
    lnorm(HBUF, F(lb + 17), F(lb + 16), XNh);
    gemm(XNh, s_wqt, F(lb + 22), nullptr, Qh, NROWS, 512, 512, 512, 512, 512, 0, nullptr, 0);
    gemm(XNh, s_wkt, F(lb + 18), nullptr, Kh, NROWS, 512, 512, 512, 512, 512, 0, nullptr, 0);
    gemm(XNh, s_wvt, F(lb + 24), nullptr, Vh, NROWS, 512, 512, 512, 512, 512, 0, nullptr, 0);
    k_attn<<<agrid, 32, 0, stream>>>(Qh, Kh, Vh, Lh, AOh);
    gemm(AOh, s_wot, F(lb + 20), HBUF, nullptr, NROWS, 512, 512, 512, 512, 512, 0, nullptr, 1);

    // cross attention (ln applied to h for Q, to z for K/V, same params)
    lnorm(HBUF, F(lb + 1), F(lb + 0), XNh);
    lnorm(z,    F(lb + 1), F(lb + 0), ZNh);
    gemm(XNh, c_wqt, F(lb + 6), nullptr, Qh, NROWS, 512, 512, 512, 512, 512, 0, nullptr, 0);
    gemm(ZNh, c_wkt, F(lb + 2), nullptr, Kh, NROWS, 512, 512, 512, 512, 512, 0, nullptr, 0);
    gemm(ZNh, c_wvt, F(lb + 8), nullptr, Vh, NROWS, 512, 512, 512, 512, 512, 0, nullptr, 0);
    k_attn<<<agrid, 32, 0, stream>>>(Qh, Kh, Vh, Lh, AOh);
    gemm(AOh, c_wot, F(lb + 4), HBUF, nullptr, NROWS, 512, 512, 512, 512, 512, 0, nullptr, 1);

    // FFN
    lnorm(HBUF, F(lb + 15), F(lb + 14), XNh);
    gemm(XNh, l1t, F(lb + 10), nullptr, F1h, NROWS, 2048, 512, 512, 512, 2048, 2, nullptr, 0);
    gemm(F1h, l2t, F(lb + 12), HBUF, nullptr, NROWS, 512, 2048, 2048, 2048, 512, 0, nullptr, 1);
  }

  // decoder head + output mixing
  lnorm(HBUF, F(10), F(9), XNh);
  gemm(XNh, dec1t, F(4), nullptr, D1h, NROWS, 512, 512, 512, 512, 512, 1, F(8), 0);
  gemm(D1h, dec2t, F(6), DECf, nullptr, NROWS, 256, 512, 512, 512, 256, 0, nullptr, 0);
  k_mix<<<(NROWS * 256 + 255) / 256, 256, 0, stream>>>(DECf, motion, maskp,
                                                       (float*)d_out, NROWS * 256);
}
